// CrossAttentionFusion_29978871726474
// MI455X (gfx1250) — matrix operand order repaired and checked by reading, not compiled, not run on hardware
//
#include <hip/hip_runtime.h>
#include <hip/hip_bf16.h>

typedef __attribute__((ext_vector_type(16))) __bf16 v16bf;
typedef __attribute__((ext_vector_type(8)))  __bf16 v8bf;
typedef __attribute__((ext_vector_type(4)))  __bf16 v4bf;
typedef __attribute__((ext_vector_type(8)))  float  v8f;
typedef __attribute__((ext_vector_type(4)))  int    v4i;

#define EDIM 1024
#define BROWS 16384

// Async global->LDS DMA path (CDNA5 GLOBAL_LOAD_ASYNC_TO_LDS_B128, ASYNCcnt)
#if defined(__has_builtin)
#if __has_builtin(__builtin_amdgcn_global_load_async_to_lds_b128) && \
    __has_builtin(__builtin_amdgcn_s_wait_asynccnt)
#define USE_ASYNC_COPY 1
#endif
#endif
#ifndef USE_ASYNC_COPY
#define USE_ASYNC_COPY 0
#endif

// ---------------------------------------------------------------------------
// f32 -> bf16 conversion (vectorized 4-wide)
// ---------------------------------------------------------------------------
__global__ __launch_bounds__(256) void cvt_f32_to_bf16(
    const float* __restrict__ src, __bf16* __restrict__ dst, long n4) {
  long i = (long)blockIdx.x * blockDim.x + threadIdx.x;
  if (i < n4) {
    float4 v = ((const float4*)src)[i];
    v4bf o;
    o[0] = (__bf16)v.x; o[1] = (__bf16)v.y; o[2] = (__bf16)v.z; o[3] = (__bf16)v.w;
    *(v4bf*)(dst + 4 * i) = o;
  }
}

// ---------------------------------------------------------------------------
// Tiled WMMA bf16 GEMM:  out[M,N] = A[M,K] * W[N,K]^T + bias[N]
// Block: 256 threads (8 wave32), tile 128x128, BK=64, double-buffered LDS,
// tiles streamed with GLOBAL_LOAD_ASYNC_TO_LDS_B128 when available.
// Wave grid 4x2: each wave computes 32 rows x 64 cols = 2x4 wmma tiles.
// EPI: 0 = f32 out, 1 = bf16 out, 2 = exact-GELU then bf16 out
// ---------------------------------------------------------------------------
#define BM 128
#define BN 128
#define BK 64
#define LDS_K 72   // padded row stride (bf16 units); 144B, 16B aligned

template <int EPI>
__global__ __launch_bounds__(256) void wmma_gemm_bf16(
    const __bf16* __restrict__ A, const __bf16* __restrict__ W,
    const float* __restrict__ bias, float* __restrict__ outF,
    __bf16* __restrict__ outB, int M, int N, int K) {
  __shared__ __bf16 sA[2][BM * LDS_K];
  __shared__ __bf16 sB[2][BN * LDS_K];

  const int tid  = threadIdx.x;
  const int lane = tid & 31;
  const int wave = tid >> 5;
  const int wr   = wave >> 1;   // 0..3 : 32-row group
  const int wc   = wave & 1;    // 0..1 : 64-col group
  const int bm0  = blockIdx.y * BM;
  const int bn0  = blockIdx.x * BN;

  // cooperative global->LDS: 2 threads per row, 64B (4x 16B) each
  const int ldRow = tid >> 1;
  const int ldCol = (tid & 1) * 32;  // bf16 units
  const size_t aBase = (size_t)(bm0 + ldRow) * K + ldCol;
  const size_t bBase = (size_t)(bn0 + ldRow) * K + ldCol;
  const int lofsA = ldRow * LDS_K + ldCol;  // bf16 units into sA[buf]
  const int lofsB = ldRow * LDS_K + ldCol;

  const int nTiles = K / BK;

#if USE_ASYNC_COPY
  using gv4i = __attribute__((address_space(1))) v4i;  // global (printed __device__)
  using lv4i = __attribute__((address_space(3))) v4i;  // LDS
  auto copy_tile = [&](int t, int buf) {
    const char* ga = (const char*)(A + aBase + (size_t)t * BK);
    const char* gb = (const char*)(W + bBase + (size_t)t * BK);
    char* la = (char*)(sA[buf] + lofsA);
    char* lb = (char*)(sB[buf] + lofsB);
#pragma unroll
    for (int i = 0; i < 4; i++) {
      __builtin_amdgcn_global_load_async_to_lds_b128(
          (gv4i*)(ga + 16 * i), (lv4i*)(la + 16 * i), 0, 0);
      __builtin_amdgcn_global_load_async_to_lds_b128(
          (gv4i*)(gb + 16 * i), (lv4i*)(lb + 16 * i), 0, 0);
    }
  };
#else
  auto copy_tile = [&](int t, int buf) {
    // sync fallback: short-lived register staging, consumed immediately
    const uint4* aSrc = (const uint4*)(A + aBase + (size_t)t * BK);
    const uint4* bSrc = (const uint4*)(W + bBase + (size_t)t * BK);
    uint4 a0 = aSrc[0], a1 = aSrc[1], a2 = aSrc[2], a3 = aSrc[3];
    uint4 b0 = bSrc[0], b1 = bSrc[1], b2 = bSrc[2], b3 = bSrc[3];
    uint4* da = (uint4*)(sA[buf] + lofsA);
    uint4* db = (uint4*)(sB[buf] + lofsB);
    da[0] = a0; da[1] = a1; da[2] = a2; da[3] = a3;
    db[0] = b0; db[1] = b1; db[2] = b2; db[3] = b3;
  };
#endif

  v8f acc[2][4] = {};

  // pipeline prologue: tile 0 into LDS buf0
  copy_tile(0, 0);
#if USE_ASYNC_COPY
  __builtin_amdgcn_s_wait_asynccnt(0);
#endif
  __syncthreads();

  for (int t = 0; t < nTiles; ++t) {
    const int cur = t & 1;

#if USE_ASYNC_COPY
    // issue DMA for the next tile; overlaps with this iteration's math
    if (t + 1 < nTiles) {
      copy_tile(t + 1, cur ^ 1);
      if (t + 2 < nTiles) {  // WGP-scope speculative prefetch one more ahead
        __builtin_prefetch(A + aBase + (size_t)(t + 2) * BK, 0, 3);
        __builtin_prefetch(W + bBase + (size_t)(t + 2) * BK, 0, 3);
      }
    }
#endif

    const __bf16* __restrict__ cA = sA[cur];
    const __bf16* __restrict__ cB = sB[cur];
#pragma unroll
    for (int kk = 0; kk < BK; kk += 32) {
      const int r   = lane & 15;
      const int kbA = kk + (lane >> 4) * 8;   // A: ISA 16-bit A layout
      const int kbB = kk + (lane >> 4) * 8;   // B: same 16-bit layout as A
      v16bf af[2], bfr[4];
#pragma unroll
      for (int mi = 0; mi < 2; mi++) {
        const __bf16* p = cA + (wr * 32 + mi * 16 + r) * LDS_K + kbA;
        v8bf lo = *(const v8bf*)p;
        v8bf hi = *(const v8bf*)(p + 16);
#pragma unroll
        for (int i = 0; i < 8; i++) { af[mi][i] = lo[i]; af[mi][i + 8] = hi[i]; }
      }
#pragma unroll
      for (int ni = 0; ni < 4; ni++) {
        const __bf16* p = cB + (wc * 64 + ni * 16 + r) * LDS_K + kbB;
        v8bf lo = *(const v8bf*)p;
        v8bf hi = *(const v8bf*)(p + 16);
#pragma unroll
        for (int i = 0; i < 8; i++) { bfr[ni][i] = lo[i]; bfr[ni][i + 8] = hi[i]; }
      }
#pragma unroll
      for (int mi = 0; mi < 2; mi++)
#pragma unroll
        for (int ni = 0; ni < 4; ni++)
          acc[mi][ni] = __builtin_amdgcn_wmma_f32_16x16x32_bf16(
              false, af[mi], false, bfr[ni], (short)0, acc[mi][ni], false, false);
    }

    if (t + 1 < nTiles) {
#if USE_ASYNC_COPY
      __builtin_amdgcn_s_wait_asynccnt(0);  // DMA of tile t+1 landed in LDS
      __syncthreads();                       // visible to all waves
#else
      __syncthreads();                       // everyone done reading buf nxt
      copy_tile(t + 1, cur ^ 1);             // load+store, regs short-lived
      __syncthreads();                       // stores visible
#endif
    }
  }

  // Epilogue. C layout (ISA 7.12.2): lane -> col = lane&15, rows (lane>>4)*8 + e
  const int col = lane & 15;
  const int rb  = (lane >> 4) * 8;
#pragma unroll
  for (int mi = 0; mi < 2; mi++) {
#pragma unroll
    for (int ni = 0; ni < 4; ni++) {
      const int gn = bn0 + wc * 64 + ni * 16 + col;
      const int gm = bm0 + wr * 32 + mi * 16 + rb;
      const float bv = bias[gn];
#pragma unroll
      for (int e = 0; e < 8; e++) {
        float v = acc[mi][ni][e] + bv;
        size_t idx = (size_t)(gm + e) * N + gn;
        if constexpr (EPI == 0) {
          outF[idx] = v;
        } else if constexpr (EPI == 1) {
          outB[idx] = (__bf16)v;
        } else {
          outB[idx] = (__bf16)(0.5f * v * (1.0f + erff(v * 0.70710678118654752f)));
        }
      }
    }
  }
}

// ---------------------------------------------------------------------------
// Row LayerNorm, one block per row of E=1024.
//   y   = LN(x0 + x1) * g + be      (+ addend, if HAS_ADDEND)
//   outF = y ;  outB = bf16(y)      (if WRITE_BF16)
// ---------------------------------------------------------------------------
template <bool HAS_ADDEND, bool WRITE_BF16>
__global__ __launch_bounds__(256) void ln_kernel(
    const float* __restrict__ x0, const float* __restrict__ x1,
    const float* __restrict__ addend, const float* __restrict__ g,
    const float* __restrict__ be, float* __restrict__ outF,
    __bf16* __restrict__ outB) {
  __shared__ float red[256];
  const int tid = threadIdx.x;
  const size_t base = (size_t)blockIdx.x * EDIM;

  float xv[4];
  float s = 0.f;
#pragma unroll
  for (int i = 0; i < 4; i++) {
    int c = tid + i * 256;
    xv[i] = x0[base + c] + x1[base + c];
    s += xv[i];
  }
  red[tid] = s;
  __syncthreads();
  for (int off = 128; off > 0; off >>= 1) {
    if (tid < off) red[tid] += red[tid + off];
    __syncthreads();
  }
  const float mu = red[0] * (1.0f / EDIM);
  __syncthreads();

  float s2 = 0.f;
#pragma unroll
  for (int i = 0; i < 4; i++) { float d = xv[i] - mu; s2 += d * d; }
  red[tid] = s2;
  __syncthreads();
  for (int off = 128; off > 0; off >>= 1) {
    if (tid < off) red[tid] += red[tid + off];
    __syncthreads();
  }
  const float rstd = rsqrtf(red[0] * (1.0f / EDIM) + 1e-5f);

#pragma unroll
  for (int i = 0; i < 4; i++) {
    int c = tid + i * 256;
    float y = (xv[i] - mu) * rstd * g[c] + be[c];
    if constexpr (HAS_ADDEND) y += addend[base + c];
    outF[base + c] = y;
    if constexpr (WRITE_BF16) outB[base + c] = (__bf16)y;
  }
}

// ---------------------------------------------------------------------------
// Host orchestration
// ---------------------------------------------------------------------------
extern "C" void kernel_launch(void* const* d_in, const int* in_sizes, int n_in,
                              void* d_out, int out_size, void* d_ws, size_t ws_size,
                              hipStream_t stream) {
  (void)in_sizes; (void)n_in; (void)out_size; (void)ws_size;
  const int E = EDIM, Bn = BROWS;

  const float* img    = (const float*)d_in[0];
  const float* txt    = (const float*)d_in[1];
  const float* w_in1  = (const float*)d_in[2];
  const float* b_in1  = (const float*)d_in[3];
  const float* w_out1 = (const float*)d_in[4];
  const float* b_out1 = (const float*)d_in[5];
  const float* w_in2  = (const float*)d_in[6];
  const float* b_in2  = (const float*)d_in[7];
  const float* w_out2 = (const float*)d_in[8];
  const float* b_out2 = (const float*)d_in[9];
  const float* g1  = (const float*)d_in[10];
  const float* be1 = (const float*)d_in[11];
  const float* g2  = (const float*)d_in[12];
  const float* be2 = (const float*)d_in[13];
  const float* g3  = (const float*)d_in[14];
  const float* be3 = (const float*)d_in[15];
  const float* w_ffn1 = (const float*)d_in[16];
  const float* b_ffn1 = (const float*)d_in[17];
  const float* w_ffn2 = (const float*)d_in[18];
  const float* b_ffn2 = (const float*)d_in[19];
  float* out = (float*)d_out;

  // workspace layout
  char* ws = (char*)d_ws;
  size_t off = 0;
  auto alloc = [&](size_t bytes) -> void* {
    void* p = ws + off;
    off += (bytes + 255) & ~(size_t)255;
    return p;
  };
  __bf16* wv1b   = (__bf16*)alloc((size_t)E * E * 2);
  __bf16* wo1b   = (__bf16*)alloc((size_t)E * E * 2);
  __bf16* wv2b   = (__bf16*)alloc((size_t)E * E * 2);
  __bf16* wo2b   = (__bf16*)alloc((size_t)E * E * 2);
  __bf16* wf1b   = (__bf16*)alloc((size_t)4 * E * E * 2);
  __bf16* wf2b   = (__bf16*)alloc((size_t)E * 4 * E * 2);
  __bf16* txtb   = (__bf16*)alloc((size_t)Bn * E * 2);
  __bf16* imgb   = (__bf16*)alloc((size_t)Bn * E * 2);   // bf16(img) after LN1
  __bf16* combb  = (__bf16*)alloc((size_t)Bn * E * 2);   // bf16(combined)
  __bf16* vb     = (__bf16*)alloc((size_t)Bn * E * 2);   // v1 / v2 (reused)
  __bf16* hb     = (__bf16*)alloc((size_t)Bn * 4 * E * 2);
  float*  attF   = (float*)alloc((size_t)Bn * E * 4);    // gemm f32 out (reused)
  float*  imgF   = (float*)alloc((size_t)Bn * E * 4);    // img after LN1
  float*  combF  = out;                                   // combined lives in d_out

  auto cvt = [&](const float* s, __bf16* d, size_t n) {
    long n4 = (long)(n / 4);
    cvt_f32_to_bf16<<<dim3((unsigned)((n4 + 255) / 256)), dim3(256), 0, stream>>>(s, d, n4);
  };

  // 0) precision conversion of weights + txt activations
  cvt(w_in1 + (size_t)2 * E * E, wv1b, (size_t)E * E);
  cvt(w_out1,                    wo1b, (size_t)E * E);
  cvt(w_in2 + (size_t)2 * E * E, wv2b, (size_t)E * E);
  cvt(w_out2,                    wo2b, (size_t)E * E);
  cvt(w_ffn1,                    wf1b, (size_t)4 * E * E);
  cvt(w_ffn2,                    wf2b, (size_t)E * 4 * E);
  cvt(txt,                       txtb, (size_t)Bn * E);

  const dim3 blk(256);
  const dim3 gE(E / BN, Bn / BM);        // N = E
  const dim3 g4E(4 * E / BN, Bn / BM);   // N = 4E

  // 1) v1 = txt @ wv1^T + bv1           -> bf16
  wmma_gemm_bf16<1><<<gE, blk, 0, stream>>>(txtb, wv1b, b_in1 + 2 * E,
                                            nullptr, vb, Bn, E, E);
  // 2) img_att = v1 @ w_out1^T + b_out1 -> f32
  wmma_gemm_bf16<0><<<gE, blk, 0, stream>>>(vb, wo1b, b_out1,
                                            attF, nullptr, Bn, E, E);
  // 3) img = LN(img_feat + img_att)     -> f32 + bf16
  ln_kernel<false, true><<<dim3(Bn), blk, 0, stream>>>(img, attF, nullptr,
                                                       g1, be1, imgF, imgb);
  // 4) v2 = img @ wv2^T + bv2           -> bf16
  wmma_gemm_bf16<1><<<gE, blk, 0, stream>>>(imgb, wv2b, b_in2 + 2 * E,
                                            nullptr, vb, Bn, E, E);
  // 5) txt_att = v2 @ w_out2^T + b_out2 -> f32
  wmma_gemm_bf16<0><<<gE, blk, 0, stream>>>(vb, wo2b, b_out2,
                                            attF, nullptr, Bn, E, E);
  // 6) combined = img + LN(txt_feat + txt_att) -> f32 (d_out) + bf16
  ln_kernel<true, true><<<dim3(Bn), blk, 0, stream>>>(txt, attF, imgF,
                                                      g2, be2, combF, combb);
  // 7) h = gelu(combined @ w_ffn1^T + b_ffn1)  -> bf16 [B,4E]
  wmma_gemm_bf16<2><<<g4E, blk, 0, stream>>>(combb, wf1b, b_ffn1,
                                             nullptr, hb, Bn, 4 * E, E);
  // 8) ffn = h @ w_ffn2^T + b_ffn2       -> f32
  wmma_gemm_bf16<0><<<gE, blk, 0, stream>>>(hb, wf2b, b_ffn2,
                                            attF, nullptr, Bn, E, 4 * E);
  // 9) out = LN(combined + ffn)
  ln_kernel<false, false><<<dim3(Bn), blk, 0, stream>>>(combF, attF, nullptr,
                                                        g3, be3, out, nullptr);
}
